// RMCCell_3882650436640
// MI455X (gfx1250) — matrix operand order repaired and checked by reading, not compiled, not run on hardware
//
#include <hip/hip_runtime.h>
#include <hip/hip_bf16.h>

// ---------------- problem constants ----------------
#define MEMD  128          // MEM
#define TOTD  384          // TOT = HEADS*QKV
#define NG    256          // NGATE
#define S1    9            // SLOTS+1 tokens
#define BT    8            // batch rows per workgroup
#define RP    80           // 72 token rows padded to 5 WMMA M-tiles
#define NTHR  256          // 8 wave32

typedef __attribute__((ext_vector_type(16))) __bf16 v16bf;
typedef __attribute__((ext_vector_type(8)))  float  v8f;

union ABu { uint4 q[2]; unsigned short u[16]; v16bf v; };

__device__ __forceinline__ unsigned short f2b(float f) {
  __hip_bfloat16 h = __float2bfloat16(f);
  unsigned short r; __builtin_memcpy(&r, &h, 2); return r;
}
__device__ __forceinline__ float b2f(unsigned short s) {
  __hip_bfloat16 h; __builtin_memcpy(&h, &s, 2); return __bfloat162float(h);
}
__device__ __forceinline__ float sigm(float x) { return 1.0f / (1.0f + __expf(-x)); }

// gfx1250 async LDS fill: 16B per lane, tracked by ASYNCcnt (ISA 15.18.3 / 08_async_tensor).
// Low 32 bits of a generic pointer into LDS == LDS byte address (ISA 10.2 aperture rule).
__device__ __forceinline__ void async_ld_b128(unsigned lds_off, const void* gptr) {
  asm volatile("global_load_async_to_lds_b128 %0, %1, off"
               :: "v"(lds_off), "v"((unsigned long long)(uintptr_t)gptr)
               : "memory");
}
__device__ __forceinline__ void wait_async0() {
  asm volatile("s_wait_asynccnt 0" ::: "memory");
}

// A-fragment (16x32 bf16, ISA 7.12.2 layout): lane m=L&15, hi=L>>4.
// VGPR0-3 <- K = hi*8 .. hi*8+7 ; VGPR4-7 <- K = 16+hi*8 .. 16+hi*8+7
__device__ __forceinline__ v16bf a_frag(const unsigned short* s, int row, int hi, int ks) {
  const unsigned short* p = s + row * MEMD + ks * 32 + hi * 8;
  ABu u;
  u.q[0] = *(const uint4*)(p);        // 8 bf16, 16B aligned
  u.q[1] = *(const uint4*)(p + 16);   // +16 elements (the K+16 run)
  return u.v;
}
// B-fragment: weights pre-packed so each lane reads 32 contiguous bytes.
__device__ __forceinline__ v16bf b_frag(const unsigned short* pw, int nt, int ks, int lane) {
  const unsigned short* p = pw + (((nt * 4 + ks) * 32) + lane) * 16;
  ABu u;
  u.q[0] = *(const uint4*)(p);
  u.q[1] = *(const uint4*)(p + 8);
  return u.v;
}

__device__ __forceinline__ v8f mm_tile(const unsigned short* sA, int row, int hi, int lane,
                                       const unsigned short* pw, int nt) {
  v8f c = {};
#pragma unroll
  for (int ks = 0; ks < 4; ++ks) {
    v16bf a = a_frag(sA, row, hi, ks);
    v16bf b = b_frag(pw, nt, ks, lane);
    c = __builtin_amdgcn_wmma_f32_16x16x32_bf16(false, a, false, b, (short)0, c, false, false);
  }
  return c;
}

__device__ __forceinline__ void red16(float& a, float& b) {
  a += __shfl_xor(a, 1, 16);  b += __shfl_xor(b, 1, 16);
  a += __shfl_xor(a, 2, 16);  b += __shfl_xor(b, 2, 16);
  a += __shfl_xor(a, 4, 16);  b += __shfl_xor(b, 4, 16);
  a += __shfl_xor(a, 8, 16);  b += __shfl_xor(b, 8, 16);
}
__device__ __forceinline__ float red32(float a) {
  a += __shfl_xor(a, 16, 32);
  a += __shfl_xor(a, 8, 32);
  a += __shfl_xor(a, 4, 32);
  a += __shfl_xor(a, 2, 32);
  a += __shfl_xor(a, 1, 32);
  return a;
}

// -------- weight repack: row-major (K,N) f32 -> WMMA-B-fragment order bf16 --------
// packed[((nt*4+ks)*32 + L)*16 + i] = W[(ks*32 + (L>>4)*8 + (i&7) + (i>>3)*16)*N + nt*16 + (L&15)]
__global__ void pack_weight(const float* __restrict__ src, unsigned short* __restrict__ dst,
                            int N, int total) {
  int idx = blockIdx.x * blockDim.x + threadIdx.x;
  if (idx >= total) return;
  int i  = idx & 15;
  int L  = (idx >> 4) & 31;
  int ks = (idx >> 9) & 3;
  int nt = idx >> 11;
  int k = ks * 32 + ((L >> 4) << 3) + (i & 7) + ((i >> 3) << 4);
  int n = nt * 16 + (L & 15);
  dst[idx] = f2b(src[k * N + n]);
}

// ---------------- fully fused RMC cell ----------------
__global__ __launch_bounds__(NTHR, 2)
void rmc_fused(const float* __restrict__ inputs, const float* __restrict__ memory,
               const unsigned short* __restrict__ pw,
               const float* __restrict__ b_in,  const float* __restrict__ b_qkv,
               const float* __restrict__ lnq_g, const float* __restrict__ lnq_b,
               const float* __restrict__ ln1_g, const float* __restrict__ ln1_b,
               const float* __restrict__ b_m1,  const float* __restrict__ b_m2,
               const float* __restrict__ ln2_g, const float* __restrict__ ln2_b,
               const float* __restrict__ b_gi,  const float* __restrict__ b_gm,
               float* __restrict__ out, int Btot) {
  // ~145 KB of LDS -> two workgroups per 320 KB WGP
  __shared__ __attribute__((aligned(16))) unsigned short sMP  [RP * MEMD]; // mp = [memory;x] bf16
  __shared__ __attribute__((aligned(16))) unsigned short sQKV [RP * TOTD]; // f32 staging -> qkv bf16
  __shared__ __attribute__((aligned(16))) unsigned short sRes [RP * MEMD]; // residuals; later tanh(mem)
  __shared__ __attribute__((aligned(16))) unsigned short sMem [RP * MEMD]; // mem / mem2 (in place)
  __shared__ __attribute__((aligned(16))) unsigned short sH1  [RP * MEMD]; // mlp hidden; later gi(f32)
  __shared__ __attribute__((aligned(16))) unsigned short sXin [16 * MEMD]; // inputs tile
  __shared__ float sSum[RP];
  __shared__ float sSsq[RP];

  const int tid  = threadIdx.x;
  const int wv   = tid >> 5;
  const int lane = tid & 31;
  const int hi   = lane >> 4;
  const int lm   = lane & 15;
  const int b0   = blockIdx.x * BT;

  const unsigned short* pW_in  = pw;            // 128x128
  const unsigned short* pW_qkv = pw + 16384;    // 128x384
  const unsigned short* pW_m1  = pw + 65536;    // 128x128
  const unsigned short* pW_m2  = pw + 81920;    // 128x128
  const unsigned short* pW_gi  = pw + 98304;    // 128x256
  const unsigned short* pW_gm  = pw + 131072;   // 128x256

  // warm L2 for first GEMM weights (global_prefetch_b8)
  __builtin_prefetch((const char*)pW_in + tid * 64, 0, 0);

  // ---------------- P0a: async-stage f32 tiles into LDS, then convert -----------
  // memory tile: contiguous 32 KB at memory + b0*1024 ; inputs tile: 4 KB.
  {
    char* stage = (char*)sQKV;                        // 36 KB scratch inside sQKV
    const char* gmem = (const char*)(memory + (size_t)b0 * 1024);
    const char* gin  = (const char*)(inputs + (size_t)b0 * MEMD);
#pragma unroll
    for (int k = 0; k < 8; ++k) {                     // 2048 x 16B = memory tile
      int c = tid + NTHR * k;
      async_ld_b128((unsigned)(uintptr_t)(stage + c * 16), gmem + c * 16);
    }
    async_ld_b128((unsigned)(uintptr_t)(stage + 32768 + tid * 16), gin + tid * 16);
  }
  // overlap: zero pads & stats while async engine fills LDS
  for (int idx = tid; idx < 8 * MEMD; idx += NTHR) {
    sXin[BT * MEMD + idx] = 0;                        // x-tile pad rows 8..15
    sMP [(72 * MEMD) + idx] = 0;                      // token pad rows 72..79
    sRes[(72 * MEMD) + idx] = 0;
  }
  if (tid < RP) { sSum[tid] = 0.0f; sSsq[tid] = 0.0f; }
  wait_async0();
  __syncthreads();
  {
    const float* stg = (const float*)sQKV;
    for (int idx = tid; idx < BT * 8 * MEMD; idx += NTHR) {
      int b = idx >> 10, rest = idx & 1023;           // 8*128=1024 per batch
      int s = rest >> 7, c = rest & 127;
      sMP[(b * S1 + s) * MEMD + c] = f2b(stg[idx]);
    }
    for (int idx = tid; idx < BT * MEMD; idx += NTHR)
      sXin[idx] = f2b(stg[8192 + idx]);
  }
  __syncthreads();

  // ---------------- P0b: x = inputs @ W_in + b_in  (1 M-tile x 8 N-tiles) --------
  {
    int nt = wv;
    v8f c = mm_tile(sXin, lm, hi, lane, pW_in, nt);
    int col = nt * 16 + lm;
    float bias = b_in[col];
#pragma unroll
    for (int r = 0; r < 8; ++r) {
      int rr = r + 8 * hi;
      if (rr < BT) sMP[(rr * S1 + 8) * MEMD + col] = f2b(c[r] + bias);
    }
  }
  __syncthreads();

  // ---------------- P1: qkv = mp @ W_qkv + b_qkv ; accumulate LN stats ----------
  for (int i = 0; i < 15; ++i) {              // 120 tiles = 5 Mt x 24 Nt, 15/wave
    int u = wv + 8 * i;
    int mt = u / 24, nt = u % 24;
    v8f c = mm_tile(sMP, mt * 16 + lm, hi, lane, pW_qkv, nt);
    int col = nt * 16 + lm;
    float bias = b_qkv[col];
    float s1a[8], s2a[8];
#pragma unroll
    for (int r = 0; r < 8; ++r) {
      float v = c[r] + bias;
      int rr = mt * 16 + r + 8 * hi;
      float s1 = v, s2 = v * v;
      red16(s1, s2);
      s1a[r] = s1; s2a[r] = s2;
      sQKV[rr * TOTD + col] = f2b(v);
    }
    if (lm == 0) {                            // single exec toggle, 16 ds_add_f32
#pragma unroll
      for (int r = 0; r < 8; ++r) {
        int rr = mt * 16 + r + 8 * hi;
        atomicAdd(&sSum[rr], s1a[r]); atomicAdd(&sSsq[rr], s2a[r]);
      }
    }
  }
  __syncthreads();

  // ---------------- P2a: LayerNorm qkv in place (n = 384) -----------------------
  for (int idx = tid; idx < RP * TOTD; idx += NTHR) {
    int row = idx / TOTD, col = idx - row * TOTD;
    float m   = sSum[row] * (1.0f / TOTD);
    float var = sSsq[row] * (1.0f / TOTD) - m * m;
    float x = b2f(sQKV[idx]);
    sQKV[idx] = f2b(lnq_g[col] * (x - m) * rsqrtf(var + 1e-5f) + lnq_b[col]);
  }
  __syncthreads();
  if (tid < RP) { sSum[tid] = 0.0f; sSsq[tid] = 0.0f; }
  __syncthreads();

  // ---------------- P2b: attention (per wave: batch wv, 4 heads) + residual -----
  {
    const int b = wv;                          // 8 waves -> 8 batches
    const float scale = rsqrtf(96.0f);         // QKV^-0.5
    for (int h = 0; h < 4; ++h) {
      const int hc = h * 96;
      for (int tq = 0; tq < S1; ++tq) {
        float qd = b2f(sQKV[(b * S1 + tq) * TOTD + hc + lane]) * scale;
        float sc[S1];
#pragma unroll
        for (int tk = 0; tk < S1; ++tk) {
          float kk = b2f(sQKV[(b * S1 + tk) * TOTD + hc + 32 + lane]);
          sc[tk] = red32(qd * kk);
        }
        float mx = sc[0];
#pragma unroll
        for (int tk = 1; tk < S1; ++tk) mx = fmaxf(mx, sc[tk]);
        float den = 0.0f;
#pragma unroll
        for (int tk = 0; tk < S1; ++tk) { sc[tk] = __expf(sc[tk] - mx); den += sc[tk]; }
        float inv = 1.0f / den;
        float o = 0.0f;
#pragma unroll
        for (int tk = 0; tk < S1; ++tk)
          o += sc[tk] * b2f(sQKV[(b * S1 + tk) * TOTD + hc + 64 + lane]);
        o *= inv;
        int rr = b * S1 + tq;
        float rv = b2f(sMP[rr * MEMD + h * 32 + lane]) + o;    // residual mp + out
        sRes[rr * MEMD + h * 32 + lane] = f2b(rv);
        float s1 = red32(rv), s2 = red32(rv * rv);
        if (lane == 0) { atomicAdd(&sSum[rr], s1); atomicAdd(&sSsq[rr], s2); }
      }
    }
  }
  __syncthreads();

  // ---------------- P3: mem = LN1(resid) (n=128) --------------------------------
  for (int idx = tid; idx < RP * MEMD; idx += NTHR) {
    int row = idx >> 7, col = idx & 127;
    float m   = sSum[row] * (1.0f / MEMD);
    float var = sSsq[row] * (1.0f / MEMD) - m * m;
    float x = b2f(sRes[idx]);
    sMem[idx] = f2b(ln1_g[col] * (x - m) * rsqrtf(var + 1e-5f) + ln1_b[col]);
  }
  __syncthreads();
  if (tid < RP) { sSum[tid] = 0.0f; sSsq[tid] = 0.0f; }
  __syncthreads();

  // ---------------- P4: h1 = mem @ W_m1 + b_m1 ----------------------------------
  for (int i = 0; i < 5; ++i) {               // 40 tiles = 5 Mt x 8 Nt
    int u = wv + 8 * i;
    int mt = u >> 3, nt = u & 7;
    v8f c = mm_tile(sMem, mt * 16 + lm, hi, lane, pW_m1, nt);
    int col = nt * 16 + lm;
    float bias = b_m1[col];
#pragma unroll
    for (int r = 0; r < 8; ++r) {
      int rr = mt * 16 + r + 8 * hi;
      sH1[rr * MEMD + col] = f2b(c[r] + bias);
    }
  }
  __syncthreads();

  // ---------------- P5: resid2 = h1 @ W_m2 + b_m2 + mem ; stats -----------------
  for (int i = 0; i < 5; ++i) {
    int u = wv + 8 * i;
    int mt = u >> 3, nt = u & 7;
    v8f c = mm_tile(sH1, mt * 16 + lm, hi, lane, pW_m2, nt);
    int col = nt * 16 + lm;
    float bias = b_m2[col];
    float s1a[8], s2a[8];
#pragma unroll
    for (int r = 0; r < 8; ++r) {
      int rr = mt * 16 + r + 8 * hi;
      float v = c[r] + bias + b2f(sMem[rr * MEMD + col]);
      float s1 = v, s2 = v * v;
      red16(s1, s2);
      s1a[r] = s1; s2a[r] = s2;
      sRes[rr * MEMD + col] = f2b(v);
    }
    if (lm == 0) {
#pragma unroll
      for (int r = 0; r < 8; ++r) {
        int rr = mt * 16 + r + 8 * hi;
        atomicAdd(&sSum[rr], s1a[r]); atomicAdd(&sSsq[rr], s2a[r]);
      }
    }
  }
  __syncthreads();

  // ---------------- P6: mem2 = LN2(resid2) -> overwrite sMem --------------------
  for (int idx = tid; idx < RP * MEMD; idx += NTHR) {
    int row = idx >> 7, col = idx & 127;
    float m   = sSum[row] * (1.0f / MEMD);
    float var = sSsq[row] * (1.0f / MEMD) - m * m;
    float x = b2f(sRes[idx]);
    sMem[idx] = f2b(ln2_g[col] * (x - m) * rsqrtf(var + 1e-5f) + ln2_b[col]);
  }
  __syncthreads();

  // ------- P6.5: materialize tanh(memory) once (rows g=b*8+s) into sRes ---------
  for (int idx = tid; idx < 64 * MEMD; idx += NTHR) {
    int g = idx >> 7, c = idx & 127;
    int b = g >> 3, s = g & 7;
    sRes[idx] = f2b(tanhf(b2f(sMP[(b * S1 + s) * MEMD + c])));
  }
  // ---------------- P7a: gi = x @ W_gi + b_gi (f32 into sH1 region) -------------
  {
    float* sGi = (float*)sH1;                 // h1 is dead; 8x256 f32 fits
    int arow = (lm < BT) ? (lm * S1 + 8) : (72 + (lm - BT));   // pad rows are zero
    for (int j = 0; j < 2; ++j) {             // 16 N-tiles, 2/wave
      int nt = wv * 2 + j;
      v8f c = mm_tile(sMP, arow, hi, lane, pW_gi, nt);
      int col = nt * 16 + lm;
      float bias = b_gi[col];
#pragma unroll
      for (int r = 0; r < 8; ++r) {
        int rr = r + 8 * hi;
        if (rr < BT) sGi[rr * NG + col] = c[r] + bias;
      }
    }
  }
  __syncthreads();

  // ---------------- P7b: gm = tanh(memory) @ W_gm ; gates ; final output --------
  {
    const float* sGi = (const float*)sH1;
    for (int j = 0; j < 4; ++j) {             // 32 units = 4 Mt x 8 paired-Nt
      int u = wv * 4 + j;
      int mt = u >> 3, np = u & 7;
      v8f ci = mm_tile(sRes, mt * 16 + lm, hi, lane, pW_gm, np);      // input-gate cols
      v8f cf = mm_tile(sRes, mt * 16 + lm, hi, lane, pW_gm, np + 8);  // forget cols (+128)
      int col = np * 16 + lm;
      float bgi_ = b_gm[col];
      float bgf_ = b_gm[col + 128];
#pragma unroll
      for (int r = 0; r < 8; ++r) {
        int gg = mt * 16 + r + 8 * hi;        // 0..63 slot row
        int bb = gg >> 3, ss = gg & 7;
        float gin = sGi[bb * NG + col];
        float gfn = sGi[bb * NG + 128 + col];
        float ig = sigm(ci[r] + bgi_ + gin);          // INPUT_BIAS = 0
        float fg = sigm(cf[r] + bgf_ + gfn + 1.0f);   // FORGET_BIAS = 1
        int mrow = bb * S1 + ss;
        float m2 = b2f(sMem[mrow * MEMD + col]);      // next_mem pre-gate (mem2 slots)
        float mo = b2f(sMP [mrow * MEMD + col]);      // original memory
        float nv = ig * tanhf(m2) + fg * mo;
        size_t gb = (size_t)(b0 + bb);
        out[gb * 1024 + (size_t)ss * 128 + col] = nv;                       // output
        out[(size_t)Btot * 1024 + (gb * 8 + ss) * 128 + col] = nv;          // next_mem
      }
    }
  }
}

extern "C" void kernel_launch(void* const* d_in, const int* in_sizes, int n_in,
                              void* d_out, int out_size, void* d_ws, size_t ws_size,
                              hipStream_t stream) {
  const float* inputs = (const float*)d_in[0];
  const float* memory = (const float*)d_in[1];
  const float* W_in   = (const float*)d_in[2];
  const float* b_in   = (const float*)d_in[3];
  const float* W_qkv  = (const float*)d_in[4];
  const float* b_qkv  = (const float*)d_in[5];
  const float* lnq_g  = (const float*)d_in[6];
  const float* lnq_b  = (const float*)d_in[7];
  const float* ln1_g  = (const float*)d_in[8];
  const float* ln1_b  = (const float*)d_in[9];
  const float* W_m1   = (const float*)d_in[10];
  const float* b_m1   = (const float*)d_in[11];
  const float* W_m2   = (const float*)d_in[12];
  const float* b_m2   = (const float*)d_in[13];
  const float* ln2_g  = (const float*)d_in[14];
  const float* ln2_b  = (const float*)d_in[15];
  const float* W_gi   = (const float*)d_in[16];
  const float* b_gi   = (const float*)d_in[17];
  const float* W_gm   = (const float*)d_in[18];
  const float* b_gm   = (const float*)d_in[19];

  const int B = in_sizes[0] / MEMD;           // 32768

  // pack all weights into WMMA-B fragment order (163,840 bf16 = 320 KB of d_ws)
  unsigned short* pw = (unsigned short*)d_ws;
  pack_weight<<<(16384 + 255) / 256, 256, 0, stream>>>(W_in,  pw,          128, 16384);
  pack_weight<<<(49152 + 255) / 256, 256, 0, stream>>>(W_qkv, pw + 16384,  384, 49152);
  pack_weight<<<(16384 + 255) / 256, 256, 0, stream>>>(W_m1,  pw + 65536,  128, 16384);
  pack_weight<<<(16384 + 255) / 256, 256, 0, stream>>>(W_m2,  pw + 81920,  128, 16384);
  pack_weight<<<(32768 + 255) / 256, 256, 0, stream>>>(W_gi,  pw + 98304,  256, 32768);
  pack_weight<<<(32768 + 255) / 256, 256, 0, stream>>>(W_gm,  pw + 131072, 256, 32768);

  rmc_fused<<<B / BT, NTHR, 0, stream>>>(inputs, memory, pw,
                                         b_in, b_qkv, lnq_g, lnq_b, ln1_g, ln1_b,
                                         b_m1, b_m2, ln2_g, ln2_b, b_gi, b_gm,
                                         (float*)d_out, B);
}
